// SelfAttention_14456859918676
// MI455X (gfx1250) — compile-verified, matrix operand order checked by reading, not run on hardware
//
#include <hip/hip_runtime.h>
#include <hip/hip_bf16.h>

typedef __bf16 bf16;
typedef __attribute__((ext_vector_type(4)))  float f32x4;
typedef __attribute__((ext_vector_type(8)))  float f32x8;
typedef __attribute__((ext_vector_type(4)))  bf16  bf16x4;
typedef __attribute__((ext_vector_type(8)))  bf16  bf16x8;
typedef __attribute__((ext_vector_type(16))) bf16  bf16x16;

#define DIM   1024
#define SEQ   4096
#define BATCH 4

// ---- WMMA helpers (CDNA5 wave32, 16x16x32 bf16) -----------------------------

static __device__ __forceinline__ bf16x16 bcat(bf16x8 lo, bf16x8 hi) {
  return __builtin_shufflevector(lo, hi, 0,1,2,3,4,5,6,7,8,9,10,11,12,13,14,15);
}

// A/B fragment: per the ISA 16-bit A-matrix layout, a lane holds 8 contiguous
// K-elements at (k + 8*half) and 8 more at (k + 16 + 8*half): two b128 loads.
static __device__ __forceinline__ bf16x16 frag(const bf16* p) {
  return bcat(*(const bf16x8*)p, *(const bf16x8*)(p + 16));
}

static __device__ __forceinline__ f32x8 wmma_bf16(bf16x16 a, bf16x16 b, f32x8 c) {
  return __builtin_amdgcn_wmma_f32_16x16x32_bf16(false, a, false, b, (short)0, c,
                                                 false, false);
}

// ---- Kernel 0: fp32 -> bf16 conversion --------------------------------------

__global__ void __launch_bounds__(256)
cvt_kernel(const float* __restrict__ src, bf16* __restrict__ dst, int n) {
  int i = (blockIdx.x * 256 + threadIdx.x) * 4;
  if (i < n) {
    f32x4 v = *(const f32x4*)(src + i);
    bf16x4 o;
    o[0] = (bf16)v[0]; o[1] = (bf16)v[1]; o[2] = (bf16)v[2]; o[3] = (bf16)v[3];
    *(bf16x4*)(dst + i) = o;
  }
}

// ---- Kernel 1: fused QKV projection, y = x @ W^T ----------------------------
// grid (M/128, DIM/128, 3); 256 threads = 8 waves; wave computes 64x32.
// z==2 (V) stores transposed: Vt[b][col][token], packed b128 stores.

__global__ void __launch_bounds__(256)
proj_kernel(const bf16* __restrict__ X, const bf16* __restrict__ W3,
            bf16* __restrict__ Qo, bf16* __restrict__ Ko, bf16* __restrict__ Vt) {
  const int lane = threadIdx.x & 31;
  const int wid  = threadIdx.x >> 5;
  const int half = lane >> 4;
  const int l16  = lane & 15;
  const int which = blockIdx.z;
  const bf16* W = W3 + (size_t)which * DIM * DIM;

  const int m0 = blockIdx.x * 128 + (wid >> 2) * 64;  // token rows (4 frags)
  const int n0 = blockIdx.y * 128 + (wid & 3) * 32;   // out cols   (2 frags)

  const bf16* xa[4];
  #pragma unroll
  for (int mf = 0; mf < 4; ++mf)
    xa[mf] = X + (size_t)(m0 + mf * 16 + l16) * DIM + half * 8;
  const bf16* wb[2];
  #pragma unroll
  for (int nf = 0; nf < 2; ++nf)
    wb[nf] = W + (size_t)(n0 + nf * 16 + l16) * DIM + half * 8;

  f32x8 acc[4][2] = {};
  for (int k = 0; k < DIM; k += 32) {
    bf16x16 a[4], b[2];
    #pragma unroll
    for (int mf = 0; mf < 4; ++mf) a[mf] = frag(xa[mf] + k);
    #pragma unroll
    for (int nf = 0; nf < 2; ++nf) b[nf] = frag(wb[nf] + k);
    #pragma unroll
    for (int mf = 0; mf < 4; ++mf)
      #pragma unroll
      for (int nf = 0; nf < 2; ++nf)
        acc[mf][nf] = wmma_bf16(a[mf], b[nf], acc[mf][nf]);
  }

  if (which < 2) {
    bf16* O = (which == 0) ? Qo : Ko;
    #pragma unroll
    for (int mf = 0; mf < 4; ++mf) {
      const int row0 = m0 + mf * 16 + half * 8;
      #pragma unroll
      for (int nf = 0; nf < 2; ++nf) {
        const int col = n0 + nf * 16 + l16;
        #pragma unroll
        for (int r = 0; r < 8; ++r)
          O[(size_t)(row0 + r) * DIM + col] = (bf16)acc[mf][nf][r];
      }
    }
  } else {
    // transposed store: lane holds 8 consecutive tokens at one column
    #pragma unroll
    for (int mf = 0; mf < 4; ++mf) {
      const int row0 = m0 + mf * 16 + half * 8;
      const int bb  = row0 >> 12;           // batch
      const int tok = row0 & (SEQ - 1);
      #pragma unroll
      for (int nf = 0; nf < 2; ++nf) {
        const int col = n0 + nf * 16 + l16;
        bf16x8 v;
        #pragma unroll
        for (int r = 0; r < 8; ++r) v[r] = (bf16)acc[mf][nf][r];
        *(bf16x8*)(Vt + ((size_t)(bb * DIM + col) * SEQ) + tok) = v;
      }
    }
  }
}

// ---- Kernel 2: flash attention ----------------------------------------------
// grid (SEQ/64, BATCH); 512 threads = 16 waves. Each block: 64 query rows.
// Wave w owns output columns [64w, 64w+64). Online softmax over 64-key chunks.

__global__ void __launch_bounds__(512)
attn_kernel(const bf16* __restrict__ Q, const bf16* __restrict__ K,
            const bf16* __restrict__ Vt, float* __restrict__ out) {
  __shared__ float Sbuf[64][65];          // padded: bank-conflict-free row scan
  __shared__ bf16  Pbuf[64][72];          // padded: 144B row stride, 16B aligned
  __shared__ __align__(16) float alpha[64];

  const int tid  = threadIdx.x;
  const int lane = tid & 31, wid = tid >> 5;
  const int half = lane >> 4, l16 = lane & 15;
  const int b    = blockIdx.y;
  const int tok0 = blockIdx.x * 64;

  const bf16* Qb = Q  + (size_t)b * SEQ * DIM;
  const bf16* Kb = K  + (size_t)b * SEQ * DIM;
  const bf16* Vb = Vt + (size_t)b * DIM * SEQ;

  const int sm = wid >> 2;                // S row-frag of this wave (0..3)
  const int sn = (wid & 3) * 16;          // S col base of this wave

  const float kScale = 0.03125f;          // 1/sqrt(1024)
  const float kLog2e = 1.44269504f;

  f32x8 acc[4][4] = {};
  float mrow = -3.4e38f, lrow = 0.f;      // owned by tid<64

  const bf16* qp = Qb + (size_t)(tok0 + sm * 16 + l16) * DIM + half * 8;

  for (int key0 = 0; key0 < SEQ; key0 += 64) {
    // ---- S = Q K^T (each wave: one 16x16 fragment, k = d = 1024) ----
    f32x8 s = {};
    const bf16* kp = Kb + (size_t)(key0 + sn + l16) * DIM + half * 8;
    for (int k = 0; k < DIM; k += 32)
      s = wmma_bf16(frag(qp + k), frag(kp + k), s);
    #pragma unroll
    for (int r = 0; r < 8; ++r)
      Sbuf[sm * 16 + half * 8 + r][sn + l16] = s[r] * kScale;
    __syncthreads();

    // ---- online softmax: thread t < 64 owns row t ----
    if (tid < 64) {
      float mc = mrow;
      for (int c = 0; c < 64; ++c) mc = fmaxf(mc, Sbuf[tid][c]);
      float al = exp2f((mrow - mc) * kLog2e);
      float sum = 0.f;
      for (int c = 0; c < 64; ++c) {
        float p = exp2f((Sbuf[tid][c] - mc) * kLog2e);
        sum += p;
        Pbuf[tid][c] = (bf16)p;
      }
      lrow = lrow * al + sum;
      mrow = mc;
      alpha[tid] = al;
    }
    __syncthreads();

    // ---- rescale accumulators by alpha[row] ----
    #pragma unroll
    for (int mf = 0; mf < 4; ++mf) {
      const f32x4* ap = (const f32x4*)&alpha[mf * 16 + half * 8];
      f32x4 a0 = ap[0], a1 = ap[1];
      #pragma unroll
      for (int nf = 0; nf < 4; ++nf)
        #pragma unroll
        for (int r = 0; r < 4; ++r) {
          acc[mf][nf][r]     *= a0[r];
          acc[mf][nf][r + 4] *= a1[r];
        }
    }

    // ---- acc += P @ V  (A from LDS, B from transposed Vt in global) ----
    #pragma unroll
    for (int ks = 0; ks < 2; ++ks) {
      bf16x16 pa[4];
      #pragma unroll
      for (int mf = 0; mf < 4; ++mf)
        pa[mf] = frag(&Pbuf[mf * 16 + l16][ks * 32 + half * 8]);
      #pragma unroll
      for (int nf = 0; nf < 4; ++nf) {
        const int col = wid * 64 + nf * 16 + l16;
        bf16x16 vb = frag(Vb + (size_t)col * SEQ + key0 + ks * 32 + half * 8);
        #pragma unroll
        for (int mf = 0; mf < 4; ++mf)
          acc[mf][nf] = wmma_bf16(pa[mf], vb, acc[mf][nf]);
      }
    }
    // next iteration's Sbuf writes don't alias Pbuf/alpha; softmax of the
    // next chunk (which rewrites Pbuf/alpha) sits behind its own barrier.
  }

  // ---- finalize: divide by l, store fp32 ----
  if (tid < 64) alpha[tid] = 1.f / lrow;
  __syncthreads();
  #pragma unroll
  for (int mf = 0; mf < 4; ++mf) {
    const f32x4* ap = (const f32x4*)&alpha[mf * 16 + half * 8];
    f32x4 a0 = ap[0], a1 = ap[1];
    const int row0 = tok0 + mf * 16 + half * 8;
    #pragma unroll
    for (int nf = 0; nf < 4; ++nf) {
      const int col = wid * 64 + nf * 16 + l16;
      #pragma unroll
      for (int r = 0; r < 8; ++r) {
        const float sc = (r < 4) ? a0[r] : a1[r - 4];
        out[((size_t)b * SEQ + row0 + r) * DIM + col] = acc[mf][nf][r] * sc;
      }
    }
  }
}

// ---- host launch ------------------------------------------------------------

extern "C" void kernel_launch(void* const* d_in, const int* in_sizes, int n_in,
                              void* d_out, int out_size, void* d_ws, size_t ws_size,
                              hipStream_t stream) {
  const float* x  = (const float*)d_in[0];
  const float* Wq = (const float*)d_in[1];
  const float* Wk = (const float*)d_in[2];
  const float* Wv = (const float*)d_in[3];

  const size_t NX = (size_t)BATCH * SEQ * DIM;  // 16M elems
  const size_t NW = (size_t)DIM * DIM;          // 1M elems

  // workspace layout (bf16): x | Wq,Wk,Wv | Q | K | Vt   (~134 MB)
  bf16* xbf = (bf16*)d_ws;
  bf16* wbf = xbf + NX;
  bf16* Qb  = wbf + 3 * NW;
  bf16* Kb  = Qb + NX;
  bf16* Vtb = Kb + NX;

  cvt_kernel<<<(unsigned)(NX / 1024), 256, 0, stream>>>(x, xbf, (int)NX);
  cvt_kernel<<<(unsigned)(NW / 1024), 256, 0, stream>>>(Wq, wbf,          (int)NW);
  cvt_kernel<<<(unsigned)(NW / 1024), 256, 0, stream>>>(Wk, wbf + NW,     (int)NW);
  cvt_kernel<<<(unsigned)(NW / 1024), 256, 0, stream>>>(Wv, wbf + 2 * NW, (int)NW);

  proj_kernel<<<dim3((BATCH * SEQ) / 128, DIM / 128, 3), 256, 0, stream>>>(
      xbf, wbf, Qb, Kb, Vtb);

  attn_kernel<<<dim3(SEQ / 64, BATCH), 512, 0, stream>>>(
      Qb, Kb, Vtb, (float*)d_out);
}
